// TemporalSwinBlock_19310172963273
// MI455X (gfx1250) — matrix operand order repaired and checked by reading, not compiled, not run on hardware
//
#include <hip/hip_runtime.h>
#include <math.h>

typedef __attribute__((ext_vector_type(16))) _Float16 v16h;
typedef __attribute__((ext_vector_type(8)))  _Float16 v8h;
typedef __attribute__((ext_vector_type(8)))  float    v8f;
typedef __attribute__((ext_vector_type(4)))  int      v4i;

#define N_HEADS 12
#define T_LEN   1000
#define B_SZ    4
#define C_DIM   768
#define HW      64
#define NWIN    63            // ceil(1000/16) windows per (b,h)
#define MROWS   4000          // B*T tokens
#define MPAD    4096          // padded to multiple of 128 for GEMM tiles

// --------------------------------------------------------------------------
// CDNA5 async global->LDS copy (ASYNCcnt path). Probe round 2 showed the
// builtin exists with typed pointers: (addrspace(1) int4*, addrspace(3)
// int4*, imm offset, imm cpol).
// --------------------------------------------------------------------------
#if __has_builtin(__builtin_amdgcn_global_load_async_to_lds_b128)
#define HAVE_ASYNC_LDS 1
#else
#define HAVE_ASYNC_LDS 0
#endif

typedef __attribute__((address_space(1))) v4i* gv4i_p;
typedef __attribute__((address_space(3))) v4i* lv4i_p;

__device__ __forceinline__ void wait_async0() {
#if __has_builtin(__builtin_amdgcn_s_wait_asynccnt)
  __builtin_amdgcn_s_wait_asynccnt(0);
#else
  asm volatile("s_wait_asynccnt 0x0" ::: "memory");
#endif
}

#if HAVE_ASYNC_LDS
// Copy 32 contiguous bytes (16 halfs) global->LDS via two async b128 ops.
// Per ISA pseudocode the instruction offset is added to BOTH addresses.
__device__ __forceinline__ void async_cp32(const _Float16* gsrc, _Float16* ldst) {
  gv4i_p g = (gv4i_p)const_cast<_Float16*>(gsrc);
  lv4i_p l = (lv4i_p)ldst;
  __builtin_amdgcn_global_load_async_to_lds_b128(g, l, 0, 0);
  __builtin_amdgcn_global_load_async_to_lds_b128(g, l, 16, 0);
}
#endif

// ---------------------------------------------------------------------------
// Kernel 1: spatial mean over (H,W) + LayerNorm over C  ->  s_h (f16, MPAD x C)
// ---------------------------------------------------------------------------
__global__ __launch_bounds__(256)
void prep_mean_ln(const float* __restrict__ x, const float* __restrict__ g,
                  const float* __restrict__ bb, _Float16* __restrict__ s_h) {
  const int row = blockIdx.x;              // b*T + t
  const int b = row / T_LEN, t = row % T_LEN;
  const int tid = threadIdx.x;
  float m[3];
#pragma unroll
  for (int e = 0; e < 3; ++e) {
    const int c = tid + e * 256;
    const float4* p = (const float4*)(x + ((size_t)(b * C_DIM + c) * T_LEN + t) * HW);
    float s = 0.f;
#pragma unroll
    for (int j = 0; j < 16; ++j) { float4 v = p[j]; s += v.x + v.y + v.z + v.w; }
    m[e] = s * (1.f / 64.f);
  }
  __shared__ float rs[256], rq[256];
  rs[tid] = m[0] + m[1] + m[2];
  rq[tid] = m[0] * m[0] + m[1] * m[1] + m[2] * m[2];
  __syncthreads();
  for (int s = 128; s > 0; s >>= 1) {
    if (tid < s) { rs[tid] += rs[tid + s]; rq[tid] += rq[tid + s]; }
    __syncthreads();
  }
  const float mu  = rs[0] * (1.f / 768.f);
  const float var = rq[0] * (1.f / 768.f) - mu * mu;
  const float inv = rsqrtf(var + 1e-5f);
#pragma unroll
  for (int e = 0; e < 3; ++e) {
    const int c = tid + e * 256;
    const float val = (m[e] - mu) * inv * g[c] + bb[c];
    s_h[(size_t)row * C_DIM + c] = (_Float16)val;
  }
}

__global__ void zero_h(_Float16* p, int n) {
  int i = blockIdx.x * 256 + threadIdx.x;
  if (i < n) p[i] = (_Float16)0.f;
}

// Weight convert + transpose: f32 [K][N] -> f16 [N][K]
__global__ void convt(const float* __restrict__ src, _Float16* __restrict__ dst,
                      int K, int N) {
  int i = blockIdx.x * 256 + threadIdx.x;
  if (i < K * N) {
    int k = i / N, n = i % N;
    dst[(size_t)n * K + k] = (_Float16)src[i];
  }
}

// ---------------------------------------------------------------------------
// WMMA GEMM: C[M][N] = A[M][K] @ B[K][N] (+bias, opt GELU, opt residual).
// Compile-time K/N/output-mode -> branch-free epilogue, 32-bit addressing.
// Double-buffered LDS, one barrier per K-step, async global->LDS when
// available. Block tile 128x128, K-step 32, wave tile 32x64 (2x4 wmma).
// ---------------------------------------------------------------------------
template <int KDIM, int NDIM, bool GELU, bool RES, bool WF32, bool WF16>
__global__ __launch_bounds__(256)
void wmma_gemm(const _Float16* __restrict__ A, const _Float16* __restrict__ BT,
               const float* __restrict__ bias, const float* __restrict__ res,
               float* __restrict__ outF, _Float16* __restrict__ outH) {
  constexpr int LDK  = 40;                 // half stride per row (80B, 16B aligned)
  constexpr int BUF  = 128 * LDK;          // halfs per buffer
  constexpr int KT   = KDIM / 32;          // K-steps
  __shared__ _Float16 As[2 * BUF];
  __shared__ _Float16 Bs[2 * BUF];
  const int tid  = threadIdx.x;
  const int lane = tid & 31;
  const int wave = tid >> 5;
  const int wm   = wave & 3;               // wave M index 0..3
  const int wn   = wave >> 2;              // wave N index 0..1
  const int m0   = blockIdx.y * 128;
  const int n0   = blockIdx.x * 128;

  v8f acc[2][4];
#pragma unroll
  for (int i = 0; i < 2; ++i)
#pragma unroll
    for (int j = 0; j < 4; ++j)
#pragma unroll
      for (int e = 0; e < 8; ++e) acc[i][j][e] = 0.f;

  // per-thread tile-copy assignment: 16 halfs (32B) of A and of B per step
  const int lrow = tid >> 1;               // 0..127
  const int lk   = (tid & 1) * 16;         // 0 or 16 halfs
  const _Float16* ga = A  + (size_t)(m0 + lrow) * KDIM + lk;
  const _Float16* gb = BT + (size_t)(n0 + lrow) * KDIM + lk;
  const int lofs = lrow * LDK + lk;

#if HAVE_ASYNC_LDS
  // prologue: async-copy tile 0 into buffer 0
  async_cp32(ga, &As[lofs]);
  async_cp32(gb, &Bs[lofs]);
#else
  v8h pa0 = *(const v8h*)(ga);
  v8h pa1 = *(const v8h*)(ga + 8);
  v8h pb0 = *(const v8h*)(gb);
  v8h pb1 = *(const v8h*)(gb + 8);
  *(v8h*)(&As[lofs])     = pa0;
  *(v8h*)(&As[lofs + 8]) = pa1;
  *(v8h*)(&Bs[lofs])     = pb0;
  *(v8h*)(&Bs[lofs + 8]) = pb1;
#endif

#pragma unroll 1
  for (int kt = 0; kt < KT; ++kt) {
    const int cur = (kt & 1) * BUF;
    const int nxt = ((kt + 1) & 1) * BUF;
    const bool have = (kt + 1) < KT;
    const int koff = have ? (kt + 1) * 32 : 0;

#if HAVE_ASYNC_LDS
    wait_async0();                         // this wave's tile-kt chunks landed
    __syncthreads();                       // whole tile kt visible; prev reads done
    if (have) {                            // async-fill next buffer during compute
      async_cp32(ga + koff, &As[nxt + lofs]);
      async_cp32(gb + koff, &Bs[nxt + lofs]);
    }
#else
    // prefetch next tile into registers (always in-bounds; store gated)
    v8h na0 = *(const v8h*)(ga + koff);
    v8h na1 = *(const v8h*)(ga + koff + 8);
    v8h nb0 = *(const v8h*)(gb + koff);
    v8h nb1 = *(const v8h*)(gb + koff + 8);
    __syncthreads();                       // tile kt visible; prev reads done
#endif

    // A fragment: lane holds row (lane&15); K chunks of 8 at kb and kb+16
    v16h af[2], bf[4];
#pragma unroll
    for (int mf = 0; mf < 2; ++mf) {
      const int r  = wm * 32 + mf * 16 + (lane & 15);
      const int kb = (lane >> 4) * 8;
      v8h lo = *(const v8h*)(&As[cur + r * LDK + kb]);
      v8h hi = *(const v8h*)(&As[cur + r * LDK + kb + 16]);
#pragma unroll
      for (int i = 0; i < 8; ++i) { af[mf][i] = lo[i]; af[mf][i + 8] = hi[i]; }
    }
    // B fragment: lane holds col (lane&15); K chunks of 8 at kb and kb+16
#pragma unroll
    for (int nf = 0; nf < 4; ++nf) {
      const int r  = wn * 64 + nf * 16 + (lane & 15);
      const int kb = (lane >> 4) * 8;
      v8h lo = *(const v8h*)(&Bs[cur + r * LDK + kb]);
      v8h hi = *(const v8h*)(&Bs[cur + r * LDK + kb + 16]);
#pragma unroll
      for (int i = 0; i < 8; ++i) { bf[nf][i] = lo[i]; bf[nf][i + 8] = hi[i]; }
    }
#pragma unroll
    for (int mf = 0; mf < 2; ++mf)
#pragma unroll
      for (int nf = 0; nf < 4; ++nf)
        acc[mf][nf] = __builtin_amdgcn_wmma_f32_16x16x32_f16(
            false, af[mf], false, bf[nf], (short)0, acc[mf][nf], false, false);

#if !HAVE_ASYNC_LDS
    if (have) {                            // fill next buffer (others read cur)
      *(v8h*)(&As[nxt + lofs])     = na0;
      *(v8h*)(&As[nxt + lofs + 8]) = na1;
      *(v8h*)(&Bs[nxt + lofs])     = nb0;
      *(v8h*)(&Bs[nxt + lofs + 8]) = nb1;
    }
#endif
  }

  // Epilogue: C/D layout -> m = j + 8*(lane>=16), n = lane&15 (all 32-bit math)
#pragma unroll
  for (int mf = 0; mf < 2; ++mf) {
#pragma unroll
    for (int nf = 0; nf < 4; ++nf) {
      const int nn = n0 + wn * 64 + nf * 16 + (lane & 15);
      const float bv = bias[nn];
      const int mb = m0 + wm * 32 + mf * 16 + 8 * (lane >> 4);
      const int idx0 = mb * NDIM + nn;
#pragma unroll
      for (int j = 0; j < 8; ++j) {
        const int idx = idx0 + j * NDIM;
        float val = acc[mf][nf][j] + bv;
        if (GELU) val = 0.5f * val * (1.f + erff(val * 0.70710678118654752f));
        if (RES)  val += res[idx];
        if (WF32) outF[idx] = val;
        if (WF16) outH[idx] = (_Float16)val;
      }
    }
  }
}

// ---------------------------------------------------------------------------
// Windowed attention per (b, head, window): l2-norm q/k, logits/tau, softmax
// (pad keys keep logit 0 and ARE included — matches the unmasked reference),
// out = attn @ v. Tiny FLOPs; plain SIMT in f32.
// ---------------------------------------------------------------------------
__global__ __launch_bounds__(256)
void win_attn(const float* __restrict__ qkv, const float* __restrict__ tau,
              _Float16* __restrict__ attout) {
  __shared__ float q[16][64], k[16][64], v[16][64];
  __shared__ float lg[16][17];
  __shared__ float sq[16], sk[16];
  const int idx = blockIdx.x;
  const int n  = idx % NWIN;
  const int hh = (idx / NWIN) % N_HEADS;
  const int b  = idx / (NWIN * N_HEADS);
  const int tid = threadIdx.x;

#pragma unroll
  for (int e = 0; e < 4; ++e) {
    const int el = tid + e * 256;
    const int r = el >> 6, d = el & 63;
    const int token = n * 16 + r;
    float qv = 0.f, kv = 0.f, vv = 0.f;
    if (token < T_LEN) {
      const size_t base = (size_t)(b * T_LEN + token) * (3 * C_DIM) + hh * 64 + d;
      qv = qkv[base];
      kv = qkv[base + C_DIM];
      vv = qkv[base + 2 * C_DIM];
    }
    q[r][d] = qv; k[r][d] = kv; v[r][d] = vv;
  }
  __syncthreads();
  if (tid < 32) {
    const float* p = (tid < 16) ? q[tid] : k[tid - 16];
    float ss = 0.f;
#pragma unroll
    for (int d = 0; d < 64; ++d) ss += p[d] * p[d];
    const float sc = 1.f / fmaxf(sqrtf(ss), 1e-12f);
    if (tid < 16) sq[tid] = sc; else sk[tid - 16] = sc;
  }
  __syncthreads();
  {
    const int w = tid >> 4, s = tid & 15;
    float dot = 0.f;
#pragma unroll
    for (int d = 0; d < 64; ++d) dot += q[w][d] * k[s][d];
    lg[w][s] = dot * sq[w] * sk[s] / (tau[hh] + 1e-6f);
  }
  __syncthreads();
  if (tid < 16) {
    float mx = -1e30f;
    for (int s = 0; s < 16; ++s) mx = fmaxf(mx, lg[tid][s]);
    float ex[16], sum = 0.f;
    for (int s = 0; s < 16; ++s) { ex[s] = __expf(lg[tid][s] - mx); sum += ex[s]; }
    const float inv = 1.f / sum;
    for (int s = 0; s < 16; ++s) lg[tid][s] = ex[s] * inv;
  }
  __syncthreads();
#pragma unroll
  for (int e = 0; e < 4; ++e) {
    const int el = tid + e * 256;
    const int w = el >> 6, d = el & 63;
    const int token = n * 16 + w;
    if (token < T_LEN) {
      float o = 0.f;
#pragma unroll
      for (int s = 0; s < 16; ++s) o += lg[w][s] * v[s][d];
      attout[(size_t)(b * T_LEN + token) * C_DIM + hh * 64 + d] = (_Float16)o;
    }
  }
}

// ---------------------------------------------------------------------------
// Final: out[b,c,t,h,w] = x + 0.5*z[b*T+t, c], float4-vectorized over (h,w).
// ---------------------------------------------------------------------------
__global__ __launch_bounds__(256)
void final_add(const float* __restrict__ x, const float* __restrict__ z,
               float* __restrict__ out) {
  const size_t i = (size_t)blockIdx.x * 256 + threadIdx.x;  // float4 index
  const size_t grp = i >> 4;                                 // (b,c,t)
  const int t = (int)(grp % T_LEN);
  const size_t bc = grp / T_LEN;
  const int c = (int)(bc % C_DIM);
  const int b = (int)(bc / C_DIM);
  const float zv = 0.5f * z[(size_t)(b * T_LEN + t) * C_DIM + c];
  float4 xv = ((const float4*)x)[i];
  xv.x += zv; xv.y += zv; xv.z += zv; xv.w += zv;
  ((float4*)out)[i] = xv;
}

// ---------------------------------------------------------------------------
extern "C" void kernel_launch(void* const* d_in, const int* in_sizes, int n_in,
                              void* d_out, int out_size, void* d_ws, size_t ws_size,
                              hipStream_t stream) {
  (void)in_sizes; (void)n_in; (void)out_size; (void)ws_size;
  const float* x      = (const float*)d_in[0];
  const float* ln_g   = (const float*)d_in[1];
  const float* ln_b   = (const float*)d_in[2];
  const float* tau    = (const float*)d_in[3];
  const float* w_qkv  = (const float*)d_in[4];
  const float* b_qkv  = (const float*)d_in[5];
  const float* w_proj = (const float*)d_in[6];
  const float* b_proj = (const float*)d_in[7];
  const float* w_mlp1 = (const float*)d_in[8];
  const float* b_mlp1 = (const float*)d_in[9];
  const float* w_mlp2 = (const float*)d_in[10];
  const float* b_mlp2 = (const float*)d_in[11];
  float* out = (float*)d_out;

  char* ws = (char*)d_ws;
  size_t off = 0;
  auto alloc = [&](size_t bytes) -> void* {
    void* p = ws + off;
    off = (off + bytes + 255) & ~(size_t)255;
    return p;
  };
  _Float16* s_h    = (_Float16*)alloc((size_t)MPAD * 768 * 2);
  float*    qkv    = (float*)   alloc((size_t)MPAD * 2304 * 4);
  _Float16* att_h  = (_Float16*)alloc((size_t)MPAD * 768 * 2);
  float*    y      = (float*)   alloc((size_t)MPAD * 768 * 4);
  _Float16* y_h    = (_Float16*)alloc((size_t)MPAD * 768 * 2);
  float*    z      = (float*)   alloc((size_t)MPAD * 768 * 4);
  _Float16* wqkvT  = (_Float16*)alloc((size_t)2304 * 768 * 2);
  _Float16* wprojT = (_Float16*)alloc((size_t)768 * 768 * 2);
  _Float16* wm1T   = (_Float16*)alloc((size_t)3072 * 768 * 2);
  _Float16* wm2T   = (_Float16*)alloc((size_t)768 * 3072 * 2);
  _Float16* hid_h  = (_Float16*)qkv;   // reuse qkv region (dead after attention)

  // Weights: f32 -> f16 transposed (one-time per launch, ~28 MB traffic)
  convt<<<(768 * 2304 + 255) / 256, 256, 0, stream>>>(w_qkv,  wqkvT,  768, 2304);
  convt<<<(768 * 768  + 255) / 256, 256, 0, stream>>>(w_proj, wprojT, 768, 768);
  convt<<<(768 * 3072 + 255) / 256, 256, 0, stream>>>(w_mlp1, wm1T,   768, 3072);
  convt<<<(3072 * 768 + 255) / 256, 256, 0, stream>>>(w_mlp2, wm2T,   3072, 768);

  // Pass 1 over x: fused spatial mean + LayerNorm
  prep_mean_ln<<<MROWS, 256, 0, stream>>>(x, ln_g, ln_b, s_h);

  // Zero GEMM pad rows
  const int padN = (MPAD - MROWS) * 768;
  zero_h<<<(padN + 255) / 256, 256, 0, stream>>>(s_h   + (size_t)MROWS * 768, padN);
  zero_h<<<(padN + 255) / 256, 256, 0, stream>>>(att_h + (size_t)MROWS * 768, padN);

  // qkv = s @ Wqkv + b   (f32 out for attention precision)
  wmma_gemm<768, 2304, false, false, true, false>
      <<<dim3(2304 / 128, MPAD / 128), 256, 0, stream>>>(
      s_h, wqkvT, b_qkv, nullptr, qkv, nullptr);

  // windowed cosine attention
  win_attn<<<B_SZ * N_HEADS * NWIN, 256, 0, stream>>>(qkv, tau, att_h);

  // y = att @ Wproj + b   (f32 for residual, f16 for next GEMM)
  wmma_gemm<768, 768, false, false, true, true>
      <<<dim3(768 / 128, MPAD / 128), 256, 0, stream>>>(
      att_h, wprojT, b_proj, nullptr, y, y_h);

  // hid = gelu(y @ Wmlp1 + b)
  wmma_gemm<768, 3072, true, false, false, true>
      <<<dim3(3072 / 128, MPAD / 128), 256, 0, stream>>>(
      y_h, wm1T, b_mlp1, nullptr, nullptr, hid_h);

  // z = y + (hid @ Wmlp2 + b)
  wmma_gemm<3072, 768, false, true, true, false>
      <<<dim3(768 / 128, MPAD / 128), 256, 0, stream>>>(
      hid_h, wm2T, b_mlp2, y, z, nullptr);

  // Pass 2 over x: out = x + 0.5*z broadcast over (H,W)
  final_add<<<(B_SZ * C_DIM * T_LEN * 16) / 256, 256, 0, stream>>>(x, z, out);
}
